// SimpleRNN_6871947674295
// MI455X (gfx1250) — compile-verified, matrix operand order checked by reading, not run on hardware
//
#include <hip/hip_runtime.h>
#include <hip/hip_bf16.h>

typedef __bf16 bf16_t;
typedef __attribute__((ext_vector_type(16))) __bf16 v16bf;
typedef __attribute__((ext_vector_type(8)))  __bf16 v8bf;
typedef __attribute__((ext_vector_type(8)))  float  v8f;

#define WMMA_BF16(a, b, c) \
  __builtin_amdgcn_wmma_f32_16x16x32_bf16(false, (a), false, (b), (short)0, (c), false, false)

static __device__ inline v16bf cat8(v8bf lo, v8bf hi) {
  return __builtin_shufflevector(lo, hi, 0,1,2,3,4,5,6,7,8,9,10,11,12,13,14,15);
}
static __device__ inline v8bf ldg8bf(const bf16_t* p) {
  return *reinterpret_cast<const v8bf*>(p);
}
static __device__ inline v8bf cvt8(const float* p) {
  v8f f = *reinterpret_cast<const v8f*>(p);
  v8bf r;
#pragma unroll
  for (int i = 0; i < 8; ++i) r[i] = (bf16_t)f[i];
  return r;
}

#define HD   512
#define TT   256
#define BB   64
#define ROWS (BB * TT)   // 16384

#define NBG  4           // batch groups (16 rows each)
#define NCG  8           // column groups (64 cols each)
#define NWG  (NBG * NCG) // 32 persistent workgroups in the recurrence

// ---------------------------------------------------------------------------
// Transpose + f32->bf16 convert the 4 square weight matrices: WT[n*512+k] = W[k*512+n]
__global__ void k_prep(const float* __restrict__ Wx1, const float* __restrict__ Wh1,
                       const float* __restrict__ Wx2, const float* __restrict__ Wh2,
                       bf16_t* __restrict__ wt) {
  const float* W = (blockIdx.y == 0) ? Wx1 : (blockIdx.y == 1) ? Wh1
                 : (blockIdx.y == 2) ? Wx2 : Wh2;
  bf16_t* T = wt + (size_t)blockIdx.y * HD * HD;
  int idx = blockIdx.x * 256 + threadIdx.x;
  int n = idx >> 9, k = idx & (HD - 1);
  T[n * HD + k] = (bf16_t)W[k * HD + n];
}

// ---------------------------------------------------------------------------
// Out[16384][512] = gather(A) @ WT^T + bias   (WT is bf16 [N=512][K=512])
__global__ __launch_bounds__(256) void k_gemm(
    const float* __restrict__ Asrc, const int* __restrict__ tokens,
    const bf16_t* __restrict__ WT, const float* __restrict__ bias,
    float* __restrict__ Out) {
  const int wave = threadIdx.x >> 5, lane = threadIdx.x & 31;
  const int lr = lane & 15, half = lane >> 4;
  const int m0 = blockIdx.x * 128 + wave * 16;
  const int cb = blockIdx.y * 64;

  const int m = m0 + lr;
  const float* arow = tokens ? (Asrc + (size_t)tokens[m] * HD)
                             : (Asrc + (size_t)m * HD);

  v8f acc[4] = {};
  for (int kk = 0; kk < HD; kk += 32) {
    const int ka = kk + half * 8;
    v16bf A = cat8(cvt8(arow + ka), cvt8(arow + ka + 16));
#pragma unroll
    for (int j = 0; j < 4; ++j) {
      const bf16_t* bp = WT + (size_t)(cb + j * 16 + lr) * HD + ka;
      acc[j] = WMMA_BF16(A, cat8(ldg8bf(bp), ldg8bf(bp + 16)), acc[j]);
    }
  }
#pragma unroll
  for (int j = 0; j < 4; ++j) {
    const int n = cb + j * 16 + lr;
    const float bn = bias[n];
#pragma unroll
    for (int g = 0; g < 8; ++g) {
      const int mm = m0 + half * 8 + g;
      Out[(size_t)mm * HD + n] = acc[j][g] + bn;
    }
  }
}

// ---------------------------------------------------------------------------
// Global sense-style barrier: bar[0]=arrive count, bar[1]=generation.
__global__ void k_barinit(unsigned* bar) { bar[0] = 0u; bar[1] = 0u; }

static __device__ inline void gbar(unsigned* bar, unsigned target) {
  __threadfence();          // release this wg's global stores
  __syncthreads();
  if (threadIdx.x == 0) {
    unsigned prev = __hip_atomic_fetch_add(&bar[0], 1u, __ATOMIC_ACQ_REL,
                                           __HIP_MEMORY_SCOPE_AGENT);
    if (prev == (unsigned)NWG - 1u) {
      __hip_atomic_store(&bar[0], 0u, __ATOMIC_RELAXED, __HIP_MEMORY_SCOPE_AGENT);
      __hip_atomic_fetch_add(&bar[1], 1u, __ATOMIC_ACQ_REL, __HIP_MEMORY_SCOPE_AGENT);
    } else {
      while (__hip_atomic_load(&bar[1], __ATOMIC_ACQUIRE,
                               __HIP_MEMORY_SCOPE_AGENT) < target) {
        __builtin_amdgcn_s_sleep(2);
      }
    }
  }
  __syncthreads();
}

// ---------------------------------------------------------------------------
// Persistent recurrence: grid (NCG, NBG), 256 threads (8 waves).
// Workgroup (cg,bg) owns output columns [cg*64, cg*64+64) of batch rows
// [bg*16, bg*16+16).  Its Wh slice (64x512 bf16 = 64KB) lives in LDS for the
// whole kernel.  h state is exchanged through a double-buffered global bf16
// buffer hx[bg][2][16][512]; one global barrier per timestep.
// Waves: n-tile j = wave&3, split-K half ksel = wave>>2.
__global__ __launch_bounds__(256) void k_rnn(
    const float* __restrict__ xproj,   // [B*T][512], bias pre-added
    const bf16_t* __restrict__ WhT,    // [N=512][K=512] bf16
    float* __restrict__ seq,           // [B*T][512]
    bf16_t* __restrict__ hx,           // [NBG][2][16][512] bf16
    unsigned* __restrict__ bar) {
  __shared__ __align__(16) bf16_t whs[64 * HD];      // 64 KB weight slice
  __shared__ __align__(16) bf16_t hslab[16 * HD];    // 16 KB h_{t-1}
  __shared__ __align__(16) float  cpart[4 * 32 * 8]; //  4 KB split-K partials

  const int tid = threadIdx.x;
  const int wave = tid >> 5, lane = tid & 31;
  const int lr = lane & 15, half = lane >> 4;
  const int j = wave & 3;
  const int ksel = wave >> 2;
  const int cg = blockIdx.x, bg = blockIdx.y;
  const int brow0 = bg * 16;
  const int nbase = cg * 64;

  bf16_t* hx_bg = hx + (size_t)bg * 2 * 16 * HD;

  // Load our 64-column Wh slice into LDS once (rows nbase..nbase+63 of WhT
  // are one contiguous 64 KB region).
  {
    const uint4* src = (const uint4*)(WhT + (size_t)nbase * HD);
    uint4* dst = (uint4*)whs;
    for (int i = tid; i < 64 * HD * 2 / 16; i += 256) dst[i] = src[i];
  }
  // Zero our slice of h exchange buffer 0 (h_{-1} = 0).
  for (int i = tid; i < 16 * 64; i += 256)
    hx_bg[(i >> 6) * HD + nbase + (i & 63)] = (bf16_t)0.0f;

  unsigned barnum = 0;
  gbar(bar, ++barnum);

  for (int t = 0; t < TT; ++t) {
    // Stage full h_{t-1} slab (16x512 bf16) for this batch group into LDS.
    {
      const uint4* src = (const uint4*)(hx_bg + (size_t)(t & 1) * 16 * HD);
      uint4* dst = (uint4*)hslab;
      for (int i = tid; i < 16 * HD * 2 / 16; i += 256) dst[i] = src[i];
    }
    // Prefetch the strided xproj elements the low waves will need at writeback.
    if (ksel == 0) {
      const int n = nbase + j * 16 + lr;
#pragma unroll
      for (int g = 0; g < 8; ++g) {
        const size_t row = (size_t)(brow0 + half * 8 + g) * TT + t;
        __builtin_prefetch(&xproj[row * HD + n], 0, 1);
      }
    }
    __syncthreads();

    v8f acc = {};
    const int kb0 = ksel * 256;
#pragma unroll
    for (int kk = 0; kk < 256; kk += 32) {
      const int ka = kb0 + kk + half * 8;
      const bf16_t* ap = hslab + lr * HD + ka;
      const bf16_t* bp = whs + (j * 16 + lr) * HD + ka;
      acc = WMMA_BF16(cat8(*(const v8bf*)ap, *(const v8bf*)(ap + 16)),
                      cat8(*(const v8bf*)bp, *(const v8bf*)(bp + 16)), acc);
    }
    if (ksel == 1) *(v8f*)(cpart + (j * 32 + lane) * 8) = acc;
    __syncthreads();

    if (ksel == 0) {
      const v8f other = *(const v8f*)(cpart + (j * 32 + lane) * 8);
      const int n = nbase + j * 16 + lr;
      bf16_t* hw = hx_bg + (size_t)((t + 1) & 1) * 16 * HD;
#pragma unroll
      for (int g = 0; g < 8; ++g) {
        const int m = half * 8 + g;
        const size_t row = (size_t)(brow0 + m) * TT + t;
        const float h = tanhf(acc[g] + other[g] + xproj[row * HD + n]);
        seq[row * HD + n] = h;
        hw[m * HD + n] = (bf16_t)h;
      }
    }
    gbar(bar, ++barnum);   // h_t published; double buffer makes one barrier enough
  }
}

// ---------------------------------------------------------------------------
// logits = h_last @ Wd + bd ; softmax over C=2.
__global__ void k_head(const float* __restrict__ seq, const float* __restrict__ Wd,
                       const float* __restrict__ bd, float* __restrict__ out) {
  const int b = threadIdx.x;
  const float* h = seq + ((size_t)b * TT + (TT - 1)) * HD;
  float a0 = bd[0], a1 = bd[1];
  for (int k = 0; k < HD; ++k) {
    const float x = h[k];
    a0 += x * Wd[2 * k + 0];
    a1 += x * Wd[2 * k + 1];
  }
  const float mx = fmaxf(a0, a1);
  const float e0 = __expf(a0 - mx), e1 = __expf(a1 - mx);
  const float inv = 1.0f / (e0 + e1);
  out[2 * b + 0] = e0 * inv;
  out[2 * b + 1] = e1 * inv;
}

// ---------------------------------------------------------------------------
extern "C" void kernel_launch(void* const* d_in, const int* in_sizes, int n_in,
                              void* d_out, int out_size, void* d_ws, size_t ws_size,
                              hipStream_t stream) {
  (void)in_sizes; (void)n_in; (void)out_size; (void)ws_size;
  const int*   tokens = (const int*)  d_in[0];
  const float* emb    = (const float*)d_in[1];
  const float* Wx1    = (const float*)d_in[2];
  const float* Wh1    = (const float*)d_in[3];
  const float* b1     = (const float*)d_in[4];
  const float* Wx2    = (const float*)d_in[5];
  const float* Wh2    = (const float*)d_in[6];
  const float* b2     = (const float*)d_in[7];
  const float* Wd     = (const float*)d_in[8];
  const float* bd     = (const float*)d_in[9];

  char* ws = (char*)d_ws;
  bf16_t* wt   = (bf16_t*)ws;                       // 4 x 512x512 bf16 = 2 MB
  bf16_t* WxT1 = wt + 0 * HD * HD;
  bf16_t* WhT1 = wt + 1 * HD * HD;
  bf16_t* WxT2 = wt + 2 * HD * HD;
  bf16_t* WhT2 = wt + 3 * HD * HD;
  float* xproj = (float*)(ws + (size_t)4 * HD * HD * sizeof(bf16_t)); // 32 MB
  float* seq   = xproj + (size_t)ROWS * HD;                           // 32 MB
  bf16_t* hx   = (bf16_t*)(seq + (size_t)ROWS * HD);                  // 128 KB
  unsigned* bar = (unsigned*)((char*)hx + (size_t)NBG * 2 * 16 * HD * sizeof(bf16_t));

  // 1) weights -> bf16 transposed
  k_prep<<<dim3(HD * HD / 256, 4), 256, 0, stream>>>(Wx1, Wh1, Wx2, Wh2, wt);
  // 2) xproj1 = emb[tokens] @ Wx1 + b1
  k_gemm<<<dim3(ROWS / 128, HD / 64), 256, 0, stream>>>(emb, tokens, WxT1, b1, xproj);
  // 3) layer-1 recurrence -> seq
  k_barinit<<<1, 1, 0, stream>>>(bar);
  k_rnn<<<dim3(NCG, NBG), 256, 0, stream>>>(xproj, WhT1, seq, hx, bar);
  // 4) xproj2 = seq @ Wx2 + b2 (reuse xproj buffer)
  k_gemm<<<dim3(ROWS / 128, HD / 64), 256, 0, stream>>>(seq, nullptr, WxT2, b2, xproj);
  // 5) layer-2 recurrence (overwrite seq buffer)
  k_barinit<<<1, 1, 0, stream>>>(bar);
  k_rnn<<<dim3(NCG, NBG), 256, 0, stream>>>(xproj, WhT2, seq, hx, bar);
  // 6) head + softmax
  k_head<<<1, BB, 0, stream>>>(seq, Wd, bd, (float*)d_out);
}
